// STRoPEAttention_7395933684052
// MI455X (gfx1250) — compile-verified, matrix operand order checked by reading, not run on hardware
//
#include <hip/hip_runtime.h>

// ---------------------------------------------------------------- types ----
typedef __attribute__((ext_vector_type(16))) __bf16 v16bf;
typedef __attribute__((ext_vector_type(8)))  float  v8f;

#define B_DIM  4
#define N_DIM  1568
#define C_DIM  1024
#define H_DIM  16
#define HD_DIM 64
#define BH_DIM (B_DIM * H_DIM)        // 64
#define M_ROWS (B_DIM * N_DIM)        // 6272
#define KV_CHUNKS (N_DIM / 32)        // 49

// round-to-nearest-even fp32 -> bf16 (raw u16)
__device__ __forceinline__ unsigned short f2bf(float f) {
    unsigned int u = __float_as_uint(f);
    u = (u + 0x7FFFu + ((u >> 16) & 1u)) >> 16;
    return (unsigned short)u;
}

// Build a 16-element bf16 WMMA fragment from two 16-byte chunks.
// CDNA5 16-bit A/B layout: lane<16 holds K {0..7,16..23},
// lane>=16 holds K {8..15,24..31}.
__device__ __forceinline__ v16bf make_frag(const unsigned short* p0,
                                           const unsigned short* p1) {
    union { uint4 u[2]; v16bf v; } f;
    f.u[0] = *reinterpret_cast<const uint4*>(p0);
    f.u[1] = *reinterpret_cast<const uint4*>(p1);
    return f.v;
}

__device__ __forceinline__ v8f wmma_bf16(v16bf a, v16bf b, v8f c) {
    return __builtin_amdgcn_wmma_f32_16x16x32_bf16(
        false, a, false, b, (short)0, c, false, false);
}

// ------------------------------------------------------------ cvt kernel ----
__global__ __launch_bounds__(256) void cvt_kernel(const float4* __restrict__ src,
                                                  ushort4* __restrict__ dst,
                                                  int n4) {
    int i = blockIdx.x * blockDim.x + threadIdx.x;
    int stride = gridDim.x * blockDim.x;
    for (; i < n4; i += stride) {
        float4 f = src[i];
        ushort4 o;
        o.x = f2bf(f.x); o.y = f2bf(f.y); o.z = f2bf(f.z); o.w = f2bf(f.w);
        dst[i] = o;
    }
}

// ------------------------------------------- shared 32x64 GEMM core ----
// 2 A-fragments (32 rows) x 4 B-fragments (64 cols), K = C_DIM.
// Ping-pong double buffering via 2x k-unroll: the fragment set being
// reloaded is already dead, so no register rotation copies are needed.
struct FragSet {
    v16bf a0, a1, b[4];
};

__device__ __forceinline__ void frag_load(FragSet& f,
                                          const unsigned short* arow0,
                                          const unsigned short* arow1,
                                          const unsigned short* const brow[4],
                                          int k, int kb) {
    f.a0 = make_frag(arow0 + k + kb, arow0 + k + kb + 16);
    f.a1 = make_frag(arow1 + k + kb, arow1 + k + kb + 16);
#pragma unroll
    for (int t = 0; t < 4; ++t)
        f.b[t] = make_frag(brow[t] + k + kb, brow[t] + k + kb + 16);
}

__device__ __forceinline__ void frag_mma(const FragSet& f, v8f acc[2][4]) {
#pragma unroll
    for (int t = 0; t < 4; ++t) {
        acc[0][t] = wmma_bf16(f.a0, f.b[t], acc[0][t]);
        acc[1][t] = wmma_bf16(f.a1, f.b[t], acc[1][t]);
    }
}

__device__ __forceinline__ void gemm_core_32x64(const unsigned short* arow0,
                                                const unsigned short* arow1,
                                                const unsigned short* const brow[4],
                                                int kb, v8f acc[2][4]) {
    FragSet fA, fB;
    frag_load(fA, arow0, arow1, brow, 0, kb);
    for (int kk = 0; kk < C_DIM - 64; kk += 64) {
        frag_load(fB, arow0, arow1, brow, kk + 32, kb);
        frag_mma(fA, acc);                                // consumes fA
        frag_load(fA, arow0, arow1, brow, kk + 64, kb);   // fA regs now dead
        frag_mma(fB, acc);                                // consumes fB
    }
    frag_load(fB, arow0, arow1, brow, C_DIM - 32, kb);
    frag_mma(fA, acc);
    frag_mma(fB, acc);
}

// -------------------------------------------- QKV GEMM + fused RoPE ----
// xw:  [6272,1024] bf16, wq: [3072,1024] bf16
// qbf/kbf: [64,1568,64] RoPE applied; vt: [64,64,1568] transposed V
__global__ __launch_bounds__(256) void qkv_rope_kernel(
    const unsigned short* __restrict__ xw,
    const unsigned short* __restrict__ wq,
    unsigned short* __restrict__ qbf,
    unsigned short* __restrict__ kbf,
    unsigned short* __restrict__ vt) {

    const int wave = blockIdx.x * (blockDim.x >> 5) + (threadIdx.x >> 5);
    const int dg   = wave % 48;               // 48 groups of 64 outdims
    const int mt   = wave / 48;               // 196 groups of 32 tokens
    const int lane = threadIdx.x & 31;
    const int r    = lane & 15;
    const int kb   = (lane >> 4) * 8;

    const unsigned short* arow0 = xw + (size_t)(mt * 32 + r) * C_DIM;
    const unsigned short* arow1 = arow0 + (size_t)16 * C_DIM;
    const unsigned short* brow[4];
#pragma unroll
    for (int t = 0; t < 4; ++t)
        brow[t] = wq + (size_t)(dg * 64 + t * 16 + r) * C_DIM;

    v8f acc[2][4];
#pragma unroll
    for (int g = 0; g < 2; ++g)
#pragma unroll
        for (int t = 0; t < 4; ++t) acc[g][t] = {};

    gemm_core_32x64(arow0, arow1, brow, kb, acc);

    // epilogue: lane holds out-dim column d; 8 token rows per acc register
#pragma unroll
    for (int t = 0; t < 4; ++t) {
        const int d0  = dg * 64 + t * 16;
        const int sec = d0 >> 10;                 // 0=q 1=k 2=v
        const int h   = (d0 >> 6) & 15;
        const int e   = (d0 & 63) + r;
        const int fi  = e & 15;
        // 10000^(-fi/16) == exp2(-fi * log2(10000)/16) -> single v_exp_f32
        const float omega = exp2f(-(float)fi * 0.83048202372f);
#pragma unroll
        for (int g = 0; g < 2; ++g) {
#pragma unroll
            for (int j = 0; j < 8; ++j) {
                const int m  = j + ((lane >> 4) << 3);
                const int n  = mt * 32 + g * 16 + m;
                const int b  = n / N_DIM;
                const int nn = n - b * N_DIM;
                const float val = acc[g][t][j];
                const float pv  = __shfl_xor(val, 1, 32);  // pair partner
                if (sec == 2) {
                    vt[((size_t)((b * H_DIM + h) * HD_DIM + e)) * N_DIM + nn] = f2bf(val);
                } else {
                    const float hp  = (float)((nn % 196) / 14);
                    const float wp  = (float)(nn % 14);
                    const float pos = (e < 32) ? hp : wp;
                    const float ang = pos * omega;
                    const float cs = __cosf(ang), sn = __sinf(ang);
                    const float out = val * cs + ((e & 1) ? pv : -pv) * sn;
                    unsigned short* dst = (sec == 0) ? qbf : kbf;
                    dst[((size_t)((b * H_DIM + h) * N_DIM + nn)) * HD_DIM + e] = f2bf(out);
                }
            }
        }
    }
}

// ------------------------------------------------- flash attention ----
// One wave per (BH, 16-query tile); 32-key chunks, online softmax.
// K fragments ping-pong between two buffers via 2x chunk unroll.
__global__ __launch_bounds__(256) void attn_kernel(
    const unsigned short* __restrict__ qbf,   // [64,1568,64]
    const unsigned short* __restrict__ kbf,   // [64,1568,64]
    const unsigned short* __restrict__ vt,    // [64,64,1568]
    unsigned short* __restrict__ obf) {       // [6272,1024]

    __shared__ __align__(16) unsigned short smem[8 * 16 * 32];  // 8 waves * 1KB

    const int wave = blockIdx.x * (blockDim.x >> 5) + (threadIdx.x >> 5);
    const int qt   = wave % (N_DIM / 16);     // 98 query tiles
    const int bh   = wave / (N_DIM / 16);     // 64 (b,h) pairs
    const int lane = threadIdx.x & 31;
    const int r    = lane & 15;
    const int kb   = (lane >> 4) * 8;
    unsigned short* sm = smem + (threadIdx.x >> 5) * 512;

    const unsigned short* kbase = kbf + (size_t)bh * N_DIM * HD_DIM;
    const unsigned short* vbase = vt  + (size_t)bh * HD_DIM * N_DIM;

    // Q fragments cached in regs (16 queries x 64 dims)
    const unsigned short* qrow = qbf + ((size_t)bh * N_DIM + qt * 16 + r) * HD_DIM;
    const v16bf aq0 = make_frag(qrow + kb,      qrow + kb + 16);
    const v16bf aq1 = make_frag(qrow + 32 + kb, qrow + 32 + kb + 16);

    v8f acc[4];
#pragma unroll
    for (int c = 0; c < 4; ++c) acc[c] = {};
    float mi[8], li[8];
#pragma unroll
    for (int j = 0; j < 8; ++j) { mi[j] = -1e30f; li[j] = 0.0f; }

    const float scale = 0.125f;   // 64^-0.5

    auto load_kfrags = [&](v16bf* kf, int t) {
        const unsigned short* kr0 = kbase + (size_t)(t * 32 + r) * HD_DIM;
        const unsigned short* kr1 = kbase + (size_t)(t * 32 + 16 + r) * HD_DIM;
        kf[0] = make_frag(kr0 + kb, kr0 + kb + 16);
        kf[1] = make_frag(kr0 + 32 + kb, kr0 + 32 + kb + 16);
        kf[2] = make_frag(kr1 + kb, kr1 + kb + 16);
        kf[3] = make_frag(kr1 + 32 + kb, kr1 + 32 + kb + 16);
    };

    // one KV chunk: S wmmas (consume kcur), prefetch K[tn] into knx,
    // V loads overlap softmax, LDS relayout of P, then PV wmmas.
    auto chunk = [&](int t, const v16bf* kcur, v16bf* knx, int tn) {
        v8f s0 = {}, s1 = {};
        s0 = wmma_bf16(aq0, kcur[0], s0);
        s0 = wmma_bf16(aq1, kcur[1], s0);
        s1 = wmma_bf16(aq0, kcur[2], s1);
        s1 = wmma_bf16(aq1, kcur[3], s1);

        load_kfrags(knx, tn);                 // kcur already consumed
        __builtin_prefetch(kbase + (size_t)(tn * 32 + lane) * HD_DIM + 2048, 0, 3);

        v16bf vf[4];
#pragma unroll
        for (int c = 0; c < 4; ++c) {
            const unsigned short* vr = vbase + (size_t)(c * 16 + r) * N_DIM + t * 32;
            vf[c] = make_frag(vr + kb, vr + kb + 16);
        }

        // online softmax; rows live in 16-lane halves -> xor 1/2/4/8 reductions
#pragma unroll
        for (int j = 0; j < 8; ++j) {
            const float v0 = s0[j] * scale;
            const float v1 = s1[j] * scale;
            float rm = fmaxf(v0, v1);
#pragma unroll
            for (int off = 8; off >= 1; off >>= 1)
                rm = fmaxf(rm, __shfl_xor(rm, off, 32));
            const float mnew = fmaxf(mi[j], rm);
            const float corr = __expf(mi[j] - mnew);
            const float p0 = __expf(v0 - mnew);
            const float p1 = __expf(v1 - mnew);
            float rs = p0 + p1;
#pragma unroll
            for (int off = 8; off >= 1; off >>= 1)
                rs += __shfl_xor(rs, off, 32);
            li[j] = li[j] * corr + rs;
            mi[j] = mnew;
#pragma unroll
            for (int c = 0; c < 4; ++c) acc[c][j] *= corr;
            const int m = j + ((lane >> 4) << 3);
            sm[m * 32 + r]      = f2bf(p0);
            sm[m * 32 + 16 + r] = f2bf(p1);
        }

        const v16bf pa = make_frag(sm + r * 32 + kb, sm + r * 32 + kb + 16);
#pragma unroll
        for (int c = 0; c < 4; ++c)
            acc[c] = wmma_bf16(pa, vf[c], acc[c]);
    };

    v16bf kA[4], kB[4];
    load_kfrags(kA, 0);
    for (int t = 0; t < KV_CHUNKS - 1; t += 2) {          // chunks 0..47
        chunk(t,     kA, kB, t + 1);
        chunk(t + 1, kB, kA, (t + 2 < KV_CHUNKS) ? t + 2 : t + 1);
    }
    chunk(KV_CHUNKS - 1, kA, kB, KV_CHUNKS - 1);          // final chunk 48

    // normalize and write o in [B,N,H*hd]
    const int b = bh >> 4, h = bh & 15;
#pragma unroll
    for (int c = 0; c < 4; ++c) {
#pragma unroll
        for (int j = 0; j < 8; ++j) {
            const int m = j + ((lane >> 4) << 3);
            const float ov = acc[c][j] / li[j];
            obf[((size_t)(b * N_DIM + qt * 16 + m)) * C_DIM + h * 64 + c * 16 + r] = f2bf(ov);
        }
    }
}

// --------------------------------------------------- output projection ----
__global__ __launch_bounds__(256) void proj_kernel(
    const unsigned short* __restrict__ obf,   // [6272,1024]
    const unsigned short* __restrict__ wp,    // [1024,1024]
    const float* __restrict__ bias,           // [1024]
    float* __restrict__ out) {                // [6272,1024]

    const int wave = blockIdx.x * (blockDim.x >> 5) + (threadIdx.x >> 5);
    const int dg   = wave % 16;               // 16 groups of 64 outdims
    const int mt   = wave / 16;               // 196 groups of 32 tokens
    const int lane = threadIdx.x & 31;
    const int r    = lane & 15;
    const int kb   = (lane >> 4) * 8;

    const unsigned short* arow0 = obf + (size_t)(mt * 32 + r) * C_DIM;
    const unsigned short* arow1 = arow0 + (size_t)16 * C_DIM;
    const unsigned short* brow[4];
#pragma unroll
    for (int t = 0; t < 4; ++t)
        brow[t] = wp + (size_t)(dg * 64 + t * 16 + r) * C_DIM;

    v8f acc[2][4];
#pragma unroll
    for (int g = 0; g < 2; ++g)
#pragma unroll
        for (int t = 0; t < 4; ++t) acc[g][t] = {};

    gemm_core_32x64(arow0, arow1, brow, kb, acc);

#pragma unroll
    for (int t = 0; t < 4; ++t) {
        const int d = dg * 64 + t * 16 + r;
        const float bv = bias[d];
#pragma unroll
        for (int g = 0; g < 2; ++g) {
#pragma unroll
            for (int j = 0; j < 8; ++j) {
                const int m = j + ((lane >> 4) << 3);
                out[(size_t)(mt * 32 + g * 16 + m) * C_DIM + d] = acc[g][t][j] + bv;
            }
        }
    }
}

// ------------------------------------------------------------- launch ----
extern "C" void kernel_launch(void* const* d_in, const int* in_sizes, int n_in,
                              void* d_out, int out_size, void* d_ws, size_t ws_size,
                              hipStream_t stream) {
    (void)in_sizes; (void)n_in; (void)out_size; (void)ws_size;
    const float* x     = (const float*)d_in[0];
    const float* W_qkv = (const float*)d_in[1];
    const float* W_proj= (const float*)d_in[2];
    const float* b_proj= (const float*)d_in[3];

    const size_t n_x  = (size_t)M_ROWS * C_DIM;          // 6,422,528
    const size_t n_wq = (size_t)3 * C_DIM * C_DIM;       // 3,145,728
    const size_t n_wp = (size_t)C_DIM * C_DIM;           // 1,048,576
    const size_t n_hd = (size_t)BH_DIM * N_DIM * HD_DIM; // 6,422,528

    // workspace carve-out (256B aligned)
    char* ws = (char*)d_ws;
    size_t off = 0;
    auto carve = [&](size_t bytes) -> char* {
        size_t cur = (off + 255) & ~(size_t)255;
        off = cur + bytes;
        return ws + cur;
    };
    unsigned short* xw  = (unsigned short*)carve(n_x  * 2);
    unsigned short* wq  = (unsigned short*)carve(n_wq * 2);
    unsigned short* wp  = (unsigned short*)carve(n_wp * 2);
    unsigned short* qbf = (unsigned short*)carve(n_hd * 2);
    unsigned short* kbf = (unsigned short*)carve(n_hd * 2);
    unsigned short* vt  = (unsigned short*)carve(n_hd * 2);
    unsigned short* obf = (unsigned short*)carve(n_x  * 2);

    cvt_kernel<<<1024, 256, 0, stream>>>((const float4*)x,      (ushort4*)xw, (int)(n_x  / 4));
    cvt_kernel<<<512,  256, 0, stream>>>((const float4*)W_qkv,  (ushort4*)wq, (int)(n_wq / 4));
    cvt_kernel<<<256,  256, 0, stream>>>((const float4*)W_proj, (ushort4*)wp, (int)(n_wp / 4));

    qkv_rope_kernel<<<(196 * 48) / 8, 256, 0, stream>>>(xw, wq, qbf, kbf, vt);
    attn_kernel<<<(BH_DIM * (N_DIM / 16)) / 8, 256, 0, stream>>>(qbf, kbf, vt, obf);
    proj_kernel<<<(196 * 16) / 8, 256, 0, stream>>>(obf, wp, b_proj, (float*)d_out);
}
